// MaxPoolingAggregator_28424093564970
// MI455X (gfx1250) — matrix operand order repaired and checked by reading, not compiled, not run on hardware
//
#include <hip/hip_runtime.h>

typedef __attribute__((ext_vector_type(16))) _Float16 v16h;
typedef __attribute__((ext_vector_type(8)))  _Float16 v8h;
typedef __attribute__((ext_vector_type(4)))  _Float16 v4h;
typedef __attribute__((ext_vector_type(8)))  float    v8f;
typedef __attribute__((ext_vector_type(4)))  float    v4f;

constexpr int NN      = 100000;   // nodes
constexpr int KN      = 32;       // neighbors per node
constexpr int D       = 128;      // D_IN == D_OUT
constexpr int TILES_M = NN / 16;  // 6250, exact

// ---------------------------------------------------------------------------
// Phase 1: H = relu(F @ W^T + b)   [N,128] = [N,128]x[128,128]
// block = 256 threads = 8 waves; each wave computes one 16x128 tile of H.
// W is staged into LDS as f16 once per block (32 KB).
// ---------------------------------------------------------------------------
__global__ __launch_bounds__(256) void gemm_relu_wmma(
    const float* __restrict__ F, const float* __restrict__ W,
    const float* __restrict__ bias, float* __restrict__ H)
{
    __shared__ _Float16 Ws[D * D];   // W[n][k] as f16, row-major

    const int tid = threadIdx.x;

    // cooperative f32 -> f16 staging of W into LDS
    for (int i = tid * 4; i < D * D; i += 256 * 4) {
        v4f w = *(const v4f*)(W + i);
        v4h hw;
        hw[0] = (_Float16)w[0]; hw[1] = (_Float16)w[1];
        hw[2] = (_Float16)w[2]; hw[3] = (_Float16)w[3];
        *(v4h*)(Ws + i) = hw;
    }
    __syncthreads();

    const int wave = tid >> 5;
    const int lane = tid & 31;
    const int hl   = lane & 15;   // row / column within tile
    const int hi   = lane >> 4;   // which lane-half

    const int tile = blockIdx.x * 8 + wave;
    if (tile >= TILES_M) return;  // whole wave exits: EXEC stays all-ones

    const float* arow = F + (size_t)(tile * 16 + hl) * D;

    v8f acc[8];
    const v8f zero = {0.f, 0.f, 0.f, 0.f, 0.f, 0.f, 0.f, 0.f};
#pragma unroll
    for (int t = 0; t < 8; ++t) acc[t] = zero;

#pragma unroll
    for (int kc = 0; kc < D; kc += 32) {
        // ---- A fragment: 16x32 f16. Lane half 'hi' holds K = kc+hi*8..+7
        //      in elems 0..7 and K = kc+16+hi*8..+7 in elems 8..15.
        v16h a;
        {
            const float* p0 = arow + kc + hi * 8;
            const float* p1 = arow + kc + 16 + hi * 8;
            v4f f0 = __builtin_nontemporal_load((const v4f*)p0);
            v4f f1 = __builtin_nontemporal_load((const v4f*)(p0 + 4));
            v4f f2 = __builtin_nontemporal_load((const v4f*)p1);
            v4f f3 = __builtin_nontemporal_load((const v4f*)(p1 + 4));
#pragma unroll
            for (int j = 0; j < 4; ++j) {
                a[j]      = (_Float16)f0[j];
                a[4 + j]  = (_Float16)f1[j];
                a[8 + j]  = (_Float16)f2[j];
                a[12 + j] = (_Float16)f3[j];
            }
        }

        // ---- 8 N-tiles: B fragment 32x16, lane = column; lane half 'hi' holds
        //      K = kc+hi*8..+7 in elems 0..7 and K = kc+16+hi*8..+7 in elems 8..15.
#pragma unroll
        for (int t = 0; t < 8; ++t) {
            const _Float16* wp = &Ws[(size_t)(t * 16 + hl) * D + kc + hi * 8];
            v8h b0 = *(const v8h*)wp;
            v8h b1 = *(const v8h*)(wp + 16);
            v16h bf;
#pragma unroll
            for (int j = 0; j < 8; ++j) { bf[j] = b0[j]; bf[8 + j] = b1[j]; }

            acc[t] = __builtin_amdgcn_wmma_f32_16x16x32_f16(
                /*neg_a=*/false, a, /*neg_b=*/false, bf,
                /*c_mod=*/(short)0, acc[t],
                /*reuse_a=*/false, /*reuse_b=*/false);
        }
    }

    // ---- epilogue: bias + ReLU, scatter per C/D layout:
    //      acc[t][r] -> (M = hi*8 + r, N = t*16 + hl)
#pragma unroll
    for (int t = 0; t < 8; ++t) {
        const int n  = t * 16 + hl;
        const float bv = bias[n];
        float* ocol = H + (size_t)(tile * 16 + hi * 8) * D + n;
#pragma unroll
        for (int r = 0; r < 8; ++r) {
            float v = acc[t][r] + bv;
            ocol[(size_t)r * D] = v > 0.f ? v : 0.f;   // keep in L2 (default RT)
        }
    }
}

// ---------------------------------------------------------------------------
// Phase 2: out[n,:] = max_k H[nbr[n,k], :]
// One wave per node; each lane owns a float4 channel group (32*4 = 128).
// H (51.2 MB) is L2-resident; each row read is one coalesced 512B request.
// ---------------------------------------------------------------------------
__global__ __launch_bounds__(256) void gather_max(
    const float* __restrict__ H, const int* __restrict__ nbr,
    float* __restrict__ O)
{
    const int wave = threadIdx.x >> 5;
    const int lane = threadIdx.x & 31;
    const size_t node = (size_t)blockIdx.x * 8 + wave;
    if (node >= NN) return;

    const int myIdx = nbr[node * KN + lane];   // lane k holds neighbor k
    const int c = lane * 4;

    int idx0 = __shfl(myIdx, 0, 32);
    v4f m = *(const v4f*)(H + (size_t)idx0 * D + c);

#pragma unroll 4
    for (int k = 1; k < KN; ++k) {
        const int idx = __shfl(myIdx, k, 32);
        v4f r = *(const v4f*)(H + (size_t)idx * D + c);
        m[0] = fmaxf(m[0], r[0]);
        m[1] = fmaxf(m[1], r[1]);
        m[2] = fmaxf(m[2], r[2]);
        m[3] = fmaxf(m[3], r[3]);
    }

    __builtin_nontemporal_store(m, (v4f*)(O + node * D + c));
}

// ---------------------------------------------------------------------------
extern "C" void kernel_launch(void* const* d_in, const int* in_sizes, int n_in,
                              void* d_out, int out_size, void* d_ws, size_t ws_size,
                              hipStream_t stream) {
    (void)in_sizes; (void)n_in; (void)out_size; (void)ws_size;
    const float* F   = (const float*)d_in[0];  // features [N,128] f32
    const int*   nb  = (const int*)  d_in[1];  // neighbors [N,32] i32
    const float* W   = (const float*)d_in[2];  // W [128,128] f32
    const float* b   = (const float*)d_in[3];  // b [128] f32
    float*       H   = (float*)d_ws;           // h [N,128] f32 (51.2 MB scratch)
    float*       O   = (float*)d_out;          // out [N,128] f32

    gemm_relu_wmma<<<(TILES_M + 7) / 8, 256, 0, stream>>>(F, W, b, H);
    gather_max<<<(NN + 7) / 8, 256, 0, stream>>>(H, nb, O);
}
